// cin_1425929142278
// MI455X (gfx1250) — matrix operand order repaired and checked by reading, not compiled, not run on hardware
//
#include <hip/hip_runtime.h>

// ---------------------------------------------------------------------------
// xDeepFM CIN, fully fused per batch element, f16 WMMA (v_wmma_f32_16x16x32_f16)
//   out[b,l,d] = sum_{i,j} x0[b,i,d] * h[b,j,d] * W[i,j,l] + bias[l]
// Rows m = b*64 + d; one workgroup (8 waves) per b; 3 CIN layers chained in LDS.
// Wave mapping: 2 M-tiles x (LN/64) N-tiles per wave -> each B fragment feeds
// two WMMAs; B fragments double-buffered in registers, prefetched 1 K-chunk
// ahead so v_wmma does not stall on just-issued global loads.
// ---------------------------------------------------------------------------

typedef _Float16 v16h __attribute__((ext_vector_type(16)));
typedef float    v8f  __attribute__((ext_vector_type(8)));

union Frag {
    unsigned int u[8];
    uint4        q[2];
    v16h         h;
};

__device__ __forceinline__ unsigned int pkmul_f16(unsigned int a, unsigned int b) {
    unsigned int r;
    asm("v_pk_mul_f16 %0, %1, %2" : "=v"(r) : "v"(a), "v"(b));
    return r;
}

__device__ __forceinline__ unsigned short f2h_bits(float x) {
    _Float16 h = (_Float16)x;
    unsigned short u;
    __builtin_memcpy(&u, &h, 2);
    return u;
}

// ---------------------------------------------------------------------------
// Pack W (fp32, [F0*FKin, LN], p = i*FKin + j) into per-lane f16 B-fragment
// order for v_wmma_f32_16x16x32_f16. B fragment (32x16, KxN):
//   lane ln holds N = (ln&15); element hh (0..15) holds K = (hh&7) + 8*(ln>=16) + 16*(hh>=8)
// Dest dword index: (((i*NCHUNK + jc)*NTILES + nt)*32 + ln)*8 + v  (hh = 2v+p)
// Layer 0 pads j from 39 -> 64 with zeros (k >= FKin -> 0).
// ---------------------------------------------------------------------------
__global__ __launch_bounds__(256)
void pack_w_kernel(const float* __restrict__ W, unsigned int* __restrict__ Wpk,
                   int FKin, int NCHUNK, int LN, int total) {
    int idx = blockIdx.x * 256 + threadIdx.x;
    if (idx >= total) return;
    int v    = idx & 7;
    int ln   = (idx >> 3) & 31;
    int rest = idx >> 8;                 // (i*NCHUNK + jc)*NTILES + nt
    int NTILES = LN >> 4;
    int nt    = rest % NTILES;
    int rest2 = rest / NTILES;
    int jc    = rest2 % NCHUNK;
    int i     = rest2 / NCHUNK;
    int n  = nt * 16 + (ln & 15);
    int k0 = jc * 32 + ((2 * v) & 7) + ((ln >> 4) << 3) + ((v >> 2) << 4);
    float w0 = (k0     < FKin) ? W[(i * FKin + k0    ) * LN + n] : 0.0f;
    float w1 = (k0 + 1 < FKin) ? W[(i * FKin + k0 + 1) * LN + n] : 0.0f;
    Wpk[idx] = (unsigned int)f2h_bits(w0) | ((unsigned int)f2h_bits(w1) << 16);
}

// ---------------------------------------------------------------------------
// One CIN layer.
//   hin  : LDS, H in A-fragment swizzled order:
//          dword (( (jc*64 + m)*2 + g)*8 + v), g = lane>=16 K-subset,
//          element hh: K = (hh&7) + 8*g + 16*(hh>=8)       [ISA A layout]
//   hout : next layer's hin (written swizzled, bias included)
//   xdup : x0 scalar per (m,i), f16 duplicated into both halves
//   outred: float[LN], init 64*bias, += sum_d acc
// Waves: mg = w&1 (rows mg*32..mg*32+31), ng = w>>1 (N group of NTPW tiles).
// ---------------------------------------------------------------------------
template <int NCHUNK, int LN, bool WRITE_H>
__device__ __forceinline__
void cin_layer(int tid,
               const unsigned int* __restrict__ Wpk,
               const float* __restrict__ bias,
               const unsigned int* xdup,
               const unsigned int* hin,
               unsigned int* hout,
               float* outred,
               float* __restrict__ outSlice) {
    constexpr int NTILES = LN / 16;
    constexpr int NTPW   = NTILES / 4;   // N tiles per wave (2 M-tiles each)
    constexpr int NC     = 39 * NCHUNK;  // total K chunks (always even)

    const int ln  = tid & 31;
    const int w   = tid >> 5;
    const int mg  = w & 1;
    const int ng  = w >> 1;
    const int l15 = ln & 15;
    const int gh  = ln >> 4;
    const int m0  = mg * 32 + l15;
    const int m1  = m0 + 16;

    for (int c = tid; c < LN; c += 256) outred[c] = 64.0f * bias[c];
    __syncthreads();

    v8f acc0[NTPW] = {};
    v8f acc1[NTPW] = {};
    Frag B0[NTPW], B1[NTPW];

    const int lanepart = (ng * NTPW) * 256 + ln * 8;

    // ---- helpers ---------------------------------------------------------
    auto loadB = [&](int C, Frag* B) {
        const unsigned int* wq = Wpk + C * (NTILES * 256) + lanepart;
#pragma unroll
        for (int tt = 0; tt < NTPW; ++tt) {
            const uint4* p = (const uint4*)(wq + tt * 256);
            B[tt].q[0] = p[0];
            B[tt].q[1] = p[1];
        }
    };

    auto computeChunk = [&](int C, Frag* B) {
        const int i  = C / NCHUNK;       // NCHUNK is a power of two
        const int jc = C % NCHUNK;
        const unsigned int xi0 = xdup[m0 * 39 + i];
        const unsigned int xi1 = xdup[m1 * 39 + i];
        const uint4* h0p = (const uint4*)(hin + (((jc << 6) + m0) * 2 + gh) * 8);
        const uint4* h1p = (const uint4*)(hin + (((jc << 6) + m1) * 2 + gh) * 8);
        uint4 a0 = h0p[0], a1 = h0p[1];
        uint4 c0 = h1p[0], c1 = h1p[1];
        Frag A0, A1;
        A0.u[0] = pkmul_f16(a0.x, xi0); A0.u[1] = pkmul_f16(a0.y, xi0);
        A0.u[2] = pkmul_f16(a0.z, xi0); A0.u[3] = pkmul_f16(a0.w, xi0);
        A0.u[4] = pkmul_f16(a1.x, xi0); A0.u[5] = pkmul_f16(a1.y, xi0);
        A0.u[6] = pkmul_f16(a1.z, xi0); A0.u[7] = pkmul_f16(a1.w, xi0);
        A1.u[0] = pkmul_f16(c0.x, xi1); A1.u[1] = pkmul_f16(c0.y, xi1);
        A1.u[2] = pkmul_f16(c0.z, xi1); A1.u[3] = pkmul_f16(c0.w, xi1);
        A1.u[4] = pkmul_f16(c1.x, xi1); A1.u[5] = pkmul_f16(c1.y, xi1);
        A1.u[6] = pkmul_f16(c1.z, xi1); A1.u[7] = pkmul_f16(c1.w, xi1);
#pragma unroll
        for (int tt = 0; tt < NTPW; ++tt) {
            acc0[tt] = __builtin_amdgcn_wmma_f32_16x16x32_f16(
                false, A0.h, false, B[tt].h, (short)0, acc0[tt], false, false);
            acc1[tt] = __builtin_amdgcn_wmma_f32_16x16x32_f16(
                false, A1.h, false, B[tt].h, (short)0, acc1[tt], false, false);
        }
    };
    // ----------------------------------------------------------------------

    loadB(0, B0);
    for (int C = 0; C < NC; C += 2) {
        loadB(C + 1, B1);                       // prefetch next chunk
        computeChunk(C, B0);
        loadB((C + 2 < NC) ? (C + 2) : (NC - 1), B0);
        computeChunk(C + 1, B1);
    }

    // Epilogue: C layout -> lane ln, vgpr r: d = mtile*16 + 8*gh + r, l = nt*16+l15
#pragma unroll
    for (int tt = 0; tt < NTPW; ++tt) {
        int   nt  = ng * NTPW + tt;
        int   l   = nt * 16 + l15;
        float bsl = bias[l];
        float s   = 0.0f;
#pragma unroll
        for (int mi = 0; mi < 2; ++mi) {
#pragma unroll
            for (int r = 0; r < 8; ++r) {
                float vv = mi ? acc1[tt][r] : acc0[tt][r];
                s += vv;
                if (WRITE_H) {
                    int d  = mg * 32 + mi * 16 + gh * 8 + r;
                    int kk = l & 31, jj = l >> 5;
                    int g2 = (kk >> 3) & 1;
                    int hh = (kk & 7) + ((kk >> 4) << 3);
                    int vd = hh >> 1, p = l & 1;
                    ((unsigned short*)hout)[((((jj << 6) + d) * 2 + g2) * 8 + vd) * 2 + p] =
                        f2h_bits(vv + bsl);
                }
            }
        }
        s += __shfl_xor(s, 16, 32);            // combine the two lane halves (same l)
        if (gh == 0) atomicAdd(&outred[l], s); // sum over this wave's 32 d values
    }
    __syncthreads();
    for (int c = tid; c < LN; c += 256) outSlice[c] = outred[c];
    __syncthreads();
}

// ---------------------------------------------------------------------------
// Fused CIN: one block per batch element b (64 rows = all d slices of b).
// ---------------------------------------------------------------------------
__global__ __launch_bounds__(256)
void cin_fused_kernel(const int* __restrict__ feat_ids,
                      const float* __restrict__ emb,
                      const unsigned int* __restrict__ Wpk0, const float* __restrict__ b0,
                      const unsigned int* __restrict__ Wpk1, const float* __restrict__ b1,
                      const unsigned int* __restrict__ Wpk2, const float* __restrict__ b2,
                      float* __restrict__ out) {
    __shared__ unsigned int hswzA[4096];   // 16 KB: layer0 in (2048 dw) / layer2 in (4096 dw)
    __shared__ unsigned int hswzB[8192];   // 32 KB: layer1 in
    __shared__ unsigned int xdup[64 * 39]; // ~10 KB: x0 scalars, f16 dup in both halves
    __shared__ float outred[256];

    const int b   = blockIdx.x;
    const int tid = threadIdx.x;

    // zero padded layer-0 H (columns 39..63 must be 0)
    for (int c = tid; c < 2048; c += 256) hswzA[c] = 0;
    __syncthreads();

    // Embedding gather: x0[b,i,d] -> xdup[m=d][i] and swizzled H0 (col i) in hswzA
    for (int idx = tid; idx < 39 * 64; idx += 256) {
        int i = idx / 64, d = idx % 64;
        int id = feat_ids[b * 39 + i];
        float x = emb[id * 64 + d];
        unsigned short hx = f2h_bits(x);
        xdup[d * 39 + i] = (unsigned int)hx * 0x00010001u;
        int kk = i & 31, jj = i >> 5;
        int g  = (kk >> 3) & 1;
        int hh = (kk & 7) + ((kk >> 4) << 3);
        int vd = hh >> 1, p = i & 1;
        ((unsigned short*)hswzA)[((((jj << 6) + d) * 2 + g) * 8 + vd) * 2 + p] = hx;
    }
    __syncthreads();

    float* outb = out + b * 448;
    cin_layer<2, 256, true >(tid, Wpk0, b0, xdup, hswzA, hswzB, outred, outb);       // K=39x64,  L=256
    cin_layer<8, 128, true >(tid, Wpk1, b1, xdup, hswzB, hswzA, outred, outb + 256); // K=39x256, L=128
    cin_layer<4,  64, false>(tid, Wpk2, b2, xdup, hswzA, hswzB, outred, outb + 384); // K=39x128, L=64
}

// ---------------------------------------------------------------------------
extern "C" void kernel_launch(void* const* d_in, const int* in_sizes, int n_in,
                              void* d_out, int out_size, void* d_ws, size_t ws_size,
                              hipStream_t stream) {
    (void)in_sizes; (void)n_in; (void)out_size; (void)ws_size;
    const int*   feat_ids = (const int*)d_in[0];
    const float* emb      = (const float*)d_in[1];
    const float* W0 = (const float*)d_in[2];
    const float* b0 = (const float*)d_in[3];
    const float* W1 = (const float*)d_in[4];
    const float* b1 = (const float*)d_in[5];
    const float* W2 = (const float*)d_in[6];
    const float* b2 = (const float*)d_in[7];

    // Packed-W workspace (f16, fragment order): dwords
    const int T0 = 39 * 2 * 16 * 32 * 8;   // 319488  (FKpad=64,  L=256)
    const int T1 = 39 * 8 *  8 * 32 * 8;   // 638976  (FKpad=256, L=128)
    const int T2 = 39 * 4 *  4 * 32 * 8;   // 159744  (FKpad=128, L=64)
    unsigned int* Wpk0 = (unsigned int*)d_ws;
    unsigned int* Wpk1 = Wpk0 + T0;
    unsigned int* Wpk2 = Wpk1 + T1;

    pack_w_kernel<<<(T0 + 255) / 256, 256, 0, stream>>>(W0, Wpk0,  39, 2, 256, T0);
    pack_w_kernel<<<(T1 + 255) / 256, 256, 0, stream>>>(W1, Wpk1, 256, 8, 128, T1);
    pack_w_kernel<<<(T2 + 255) / 256, 256, 0, stream>>>(W2, Wpk2, 128, 4,  64, T2);

    cin_fused_kernel<<<2048, 256, 0, stream>>>(feat_ids, emb,
                                               Wpk0, b0, Wpk1, b1, Wpk2, b2,
                                               (float*)d_out);
}
